// SelfAttentionModule_81509889343681
// MI455X (gfx1250) — matrix operand order repaired and checked by reading, not compiled, not run on hardware
//
#include <hip/hip_runtime.h>
#include <hip/hip_bf16.h>
#include <math.h>

// ---- problem constants (match reference) ----
#define HH    56
#define WWD   56
#define HWSZ  (HH * WWD)       // 3136
#define NN    2
#define CIN   256
#define RELC  16
#define OUTC  256
#define KK    7
#define KSQ   49
#define GRP   32               // OUT / SHARE
#define SHR   8
#define TCH   18               // REL + 2 concat channels

typedef __attribute__((ext_vector_type(16))) _Float16 v16h;
typedef __attribute__((ext_vector_type(8)))  float    v8f;

__device__ __forceinline__ int reflect_idx(int v, int n) {
  if (v < 0) v = -v;
  if (v >= n) v = 2 * n - 2 - v;
  return v;
}

// =====================================================================
// Unified WMMA GEMM:  Y[n, m, s] = sum_c Wm[m,c] * f(Bsrc[n, c, s]) (+ bias[m])
//   f = identity, or fused BN (scale/shift per channel) + ReLU.
//   Column space = Ntot*S flattened; col -> (n = col/S, inner = col%S).
//   Fully templated: all bounds checks / BN / bias / address offsets fold at
//   compile time -> straight-line loads, fully unrolled k-loop, MT back-to-back
//   v_wmma_f32_16x16x32_f16 per k-step sharing one B operand.
//   Block = 128 threads = 4 wave32's, each wave owns one 16-col tile.
// =====================================================================
template<int M, int C, int MT, long S, bool HAS_BIAS, bool HAS_BN>
__global__ __launch_bounds__(128)
void gemm16_wmma(const float* __restrict__ Wm,
                 const float* __restrict__ bias,
                 const float* __restrict__ Bsrc,
                 const float* __restrict__ bnScale,
                 const float* __restrict__ bnShift,
                 float* __restrict__ Y)
{
  const int lane = threadIdx.x & 31;
  const int wave = threadIdx.x >> 5;
  const int nIdx = lane & 15;       // N index within 16x16 tile
  const int hi   = lane >> 4;       // half-wave select
  const int m0   = blockIdx.y * (16 * MT);

  const long col   = ((long)blockIdx.x * 4 + wave) * 16 + nIdx;
  const long nb    = col / S;       // S is a compile-time constant
  const long inner = col % S;
  const float* bCol = Bsrc + nb * (long)C * S + inner;   // + c*S walks channels

  v8f acc[MT];
#pragma unroll
  for (int t = 0; t < MT; ++t) acc[t] = (v8f){};

#pragma unroll
  for (int k0 = 0; k0 < C; k0 += 32) {
    // ---- B 32x16 f16 layout: lane n=l%16; half-wave hi covers K = hi*16 .. hi*16+15
    v16h b;
#pragma unroll
    for (int e = 0; e < 16; ++e) {
      const int kb = k0 + hi * 16 + e;        // compile-time bound check below
      float bv = 0.f;
      if (kb < C) {
        bv = bCol[(long)kb * S];
        if (HAS_BN) bv = fmaxf(fmaf(bv, bnScale[kb], bnShift[kb]), 0.f);
      }
      b[e] = (_Float16)bv;
    }
    // ---- MT m-tiles share this B operand
#pragma unroll
    for (int t = 0; t < MT; ++t) {
      const float* aRow = Wm + (long)(m0 + t * 16 + nIdx) * C;
      // A 16x32 f16 layout: lane m=l%16; half-wave hi covers K = hi*16 .. hi*16+15
      v16h a;
#pragma unroll
      for (int e = 0; e < 16; ++e) {
        const int ka = k0 + hi * 16 + e;
        a[e] = (ka < C) ? (_Float16)aRow[ka] : (_Float16)0.f;
      }
      acc[t] = __builtin_amdgcn_wmma_f32_16x16x32_f16(
          /*neg_a=*/false, a, /*neg_b=*/false, b,
          /*c_mod=*/(short)0, acc[t], /*reuse_a=*/false, /*reuse_b=*/false);
    }
  }

  // D layout: VGPR v, lanes 0-15 -> M=v, lanes 16-31 -> M=v+8; N = lane%16
  float* yCol = Y + nb * (long)M * S + inner;
#pragma unroll
  for (int t = 0; t < MT; ++t) {
#pragma unroll
    for (int v = 0; v < 8; ++v) {
      const int m = m0 + t * 16 + v + hi * 8;
      float r = acc[t][v];
      if (HAS_BIAS) r += bias[m];
      yCol[(long)m * S] = r;
    }
  }
}

// ---- pos = wp @ [lw; lh] + bp  ->  pos[2, HW] ----
__global__ void pos_kernel(const float* __restrict__ wp, const float* __restrict__ bp,
                           float* __restrict__ pos)
{
  int q = blockIdx.x * blockDim.x + threadIdx.x;
  if (q >= HWSZ) return;
  int h = q / WWD, w = q % WWD;
  float lw = -1.f + 2.f * (float)w / (float)(WWD - 1);
  float lh = -1.f + 2.f * (float)h / (float)(HH - 1);
  pos[q]        = fmaf(wp[0], lw, fmaf(wp[1], lh, bp[0]));
  pos[HWSZ + q] = fmaf(wp[2], lw, fmaf(wp[3], lh, bp[1]));
}

// ---- t1[n, c, k, q] = (c<16) ? x1[n,c,q]-x2[n,c,idx(k,q)] : pos[c-16,q]-pos[c-16,idx(k,q)] ----
__global__ void build_t1_kernel(const float* __restrict__ x1, const float* __restrict__ x2,
                                const float* __restrict__ pos, float* __restrict__ t1)
{
  long flat = (long)blockIdx.x * blockDim.x + threadIdx.x;
  const long total = (long)NN * TCH * KSQ * HWSZ;
  if (flat >= total) return;
  int q = (int)(flat % HWSZ);
  int k = (int)((flat / HWSZ) % KSQ);
  int c = (int)((flat / ((long)HWSZ * KSQ)) % TCH);
  int n = (int)(flat / ((long)HWSZ * KSQ * TCH));
  int h = q / WWD, w = q % WWD;
  int hh = reflect_idx(h + k / KK - 3, HH);
  int wn = reflect_idx(w + k % KK - 3, WWD);
  int qn = hh * WWD + wn;
  float val;
  if (c < RELC) {
    const float* a = x1 + ((long)n * RELC + c) * HWSZ;
    const float* b = x2 + ((long)n * RELC + c) * HWSZ;
    val = a[q] - b[qn];
  } else {
    const float* p = pos + (long)(c - RELC) * HWSZ;
    val = p[q] - p[qn];
  }
  t1[flat] = val;
}

// ---- per-channel sum / sumsq over [Ntot, C, S]; wave-reduce then f32 atomics ----
template<int C, long S>
__global__ void stats_kernel(const float* __restrict__ src, long cols,
                             float* __restrict__ sumBuf, float* __restrict__ sqBuf)
{
  long col = (long)blockIdx.x * blockDim.x + threadIdx.x;
  bool ok = col < cols;
  long nb    = ok ? col / S : 0;
  long inner = ok ? col % S : 0;
  const float* base = src + nb * (long)C * S + inner;
#pragma unroll 1
  for (int c = 0; c < C; ++c) {
    float val = ok ? base[(long)c * S] : 0.f;
    float s = val, s2 = val * val;
    for (int off = 16; off > 0; off >>= 1) {
      s  += __shfl_down(s,  off, 32);
      s2 += __shfl_down(s2, off, 32);
    }
    if ((threadIdx.x & 31) == 0) {
      atomicAdd(&sumBuf[c], s);
      atomicAdd(&sqBuf[c], s2);
    }
  }
}

// ---- BN finalize: scale = g*rsqrt(var+eps); shift = be - mean*scale ----
__global__ void bn_finalize_kernel(const float* __restrict__ sumBuf, const float* __restrict__ sqBuf,
                                   const float* __restrict__ gamma,  const float* __restrict__ beta,
                                   int C, float invCount,
                                   float* __restrict__ scale, float* __restrict__ shift)
{
  int c = threadIdx.x;
  if (c >= C) return;
  float mean = sumBuf[c] * invCount;
  float var  = sqBuf[c] * invCount - mean * mean;
  float sc   = gamma[c] * rsqrtf(var + 1e-5f);
  scale[c] = sc;
  shift[c] = beta[c] - mean * sc;
}

// ---- fused softmax(k^2) + windowed aggregation (u3 never materialized) ----
__global__ void softmax_agg_kernel(const float* __restrict__ wlog,
                                   const float* __restrict__ x3,
                                   float* __restrict__ out)
{
  long flat = (long)blockIdx.x * blockDim.x + threadIdx.x;
  const long total = (long)NN * GRP * HWSZ;
  if (flat >= total) return;
  int q = (int)(flat % HWSZ);
  int g = (int)((flat / HWSZ) % GRP);
  int n = (int)(flat / ((long)GRP * HWSZ));

  const float* wl = wlog + ((long)(n * GRP + g) * KSQ) * HWSZ + q;
  float v[KSQ];
  float mx = -1e30f;
#pragma unroll
  for (int k = 0; k < KSQ; ++k) { v[k] = wl[(long)k * HWSZ]; mx = fmaxf(mx, v[k]); }
  float s = 0.f;
#pragma unroll
  for (int k = 0; k < KSQ; ++k) { v[k] = __expf(v[k] - mx); s += v[k]; }
  float inv = 1.f / s;

  int h = q / WWD, w = q % WWD;
  const float* xb = x3 + ((long)n * OUTC + g * SHR) * HWSZ;
  float acc[SHR];
#pragma unroll
  for (int t = 0; t < SHR; ++t) acc[t] = 0.f;
#pragma unroll
  for (int k = 0; k < KSQ; ++k) {
    int hh = reflect_idx(h + k / KK - 3, HH);
    int wn = reflect_idx(w + k % KK - 3, WWD);
    int qn = hh * WWD + wn;
    float wk = v[k] * inv;
#pragma unroll
    for (int t = 0; t < SHR; ++t)
      acc[t] = fmaf(wk, xb[(long)t * HWSZ + qn], acc[t]);
  }
  float* ob = out + ((long)n * OUTC + g * SHR) * HWSZ + q;
#pragma unroll
  for (int t = 0; t < SHR; ++t) ob[(long)t * HWSZ] = acc[t];
}

__global__ void zero_kernel(float* __restrict__ p, int n)
{
  int i = blockIdx.x * blockDim.x + threadIdx.x;
  if (i < n) p[i] = 0.f;
}

// =====================================================================
extern "C" void kernel_launch(void* const* d_in, const int* in_sizes, int n_in,
                              void* d_out, int out_size, void* d_ws, size_t ws_size,
                              hipStream_t stream)
{
  const float* x   = (const float*)d_in[0];
  const float* w1  = (const float*)d_in[1];
  const float* b1  = (const float*)d_in[2];
  const float* w2  = (const float*)d_in[3];
  const float* b2  = (const float*)d_in[4];
  const float* w3  = (const float*)d_in[5];
  const float* b3  = (const float*)d_in[6];
  const float* wp  = (const float*)d_in[7];
  const float* bp  = (const float*)d_in[8];
  const float* g1  = (const float*)d_in[9];
  const float* be1 = (const float*)d_in[10];
  const float* cw1 = (const float*)d_in[11];
  const float* g2  = (const float*)d_in[12];
  const float* be2 = (const float*)d_in[13];
  const float* cw2 = (const float*)d_in[14];
  const float* cb2 = (const float*)d_in[15];
  float* out = (float*)d_out;

  // ---- workspace carve-up (floats); total ~22.1M floats (~88.4 MB) ----
  float* ws = (float*)d_ws;
  long off = 0;
  float* x1b  = ws + off; off += (long)NN * RELC * HWSZ;        // 100,352
  float* x2b  = ws + off; off += (long)NN * RELC * HWSZ;        // 100,352
  float* x3b  = ws + off; off += (long)NN * OUTC * HWSZ;        // 1,605,632
  float* posb = ws + off; off += 2L * HWSZ;                     // 6,272
  float* t1b  = ws + off; off += (long)NN * TCH * KSQ * HWSZ;   // 5,531,904
  float* t2b  = ws + off; off += (long)NN * RELC * KSQ * HWSZ;  // 4,917,248
  float* wlb  = ws + off; off += (long)NN * GRP * KSQ * HWSZ;   // 9,834,496
  float* st   = ws + off;                                       // stats block
  float* sum1 = st;        float* sq1 = st + TCH;
  float* sum2 = st + 36;   float* sq2 = st + 52;                // 36 = 2*18
  float* sc1  = st + 68;   float* sh1 = st + 86;
  float* sc2  = st + 104;  float* sh2 = st + 120;

  constexpr long S1 = HWSZ;               // per-channel spatial for x-projections
  constexpr long S2 = (long)KSQ * HWSZ;   // per-channel spatial for t tensors
  const long cols1 = (long)NN * S1;       // 6,272   (mult of 64)
  const long cols2 = (long)NN * S2;       // 307,328 (mult of 64)
  const float invCnt = 1.0f / (float)cols2;

  dim3 blk(128);  // 4 wave32's per workgroup

  // 0) zero BN accumulators (harness does not re-poison between replays)
  zero_kernel<<<1, 128, 0, stream>>>(st, 68);

  // 1) projections: x1 = w1@x+b1, x2 = w2@x+b2, x3 = w3@x+b3
  gemm16_wmma<RELC, CIN, 1, S1, true, false>
      <<<dim3(cols1 / 64, 1), blk, 0, stream>>>(w1, b1, x, nullptr, nullptr, x1b);
  gemm16_wmma<RELC, CIN, 1, S1, true, false>
      <<<dim3(cols1 / 64, 1), blk, 0, stream>>>(w2, b2, x, nullptr, nullptr, x2b);
  gemm16_wmma<OUTC, CIN, 2, S1, true, false>
      <<<dim3(cols1 / 64, OUTC / 32), blk, 0, stream>>>(w3, b3, x, nullptr, nullptr, x3b);

  // 2) position features
  pos_kernel<<<(HWSZ + 255) / 256, 256, 0, stream>>>(wp, bp, posb);

  // 3) t1 = concat(sub2, subp)
  {
    long tot = (long)NN * TCH * KSQ * HWSZ;
    build_t1_kernel<<<(unsigned)((tot + 255) / 256), 256, 0, stream>>>(x1b, x2b, posb, t1b);
  }

  // 4) BN1 stats + finalize
  stats_kernel<TCH, S2><<<(unsigned)((cols2 + 255) / 256), 256, 0, stream>>>(t1b, cols2, sum1, sq1);
  bn_finalize_kernel<<<1, 32, 0, stream>>>(sum1, sq1, g1, be1, TCH, invCnt, sc1, sh1);

  // 5) t2 = cw1 @ relu(bn1(t1))   (BN+ReLU fused into B-operand load)
  gemm16_wmma<RELC, TCH, 1, S2, false, true>
      <<<dim3(cols2 / 64, 1), blk, 0, stream>>>(cw1, nullptr, t1b, sc1, sh1, t2b);

  // 6) BN2 stats + finalize
  stats_kernel<RELC, S2><<<(unsigned)((cols2 + 255) / 256), 256, 0, stream>>>(t2b, cols2, sum2, sq2);
  bn_finalize_kernel<<<1, 32, 0, stream>>>(sum2, sq2, g2, be2, RELC, invCnt, sc2, sh2);

  // 7) wlog = cw2 @ relu(bn2(t2)) + cb2
  gemm16_wmma<GRP, RELC, 2, S2, true, true>
      <<<dim3(cols2 / 64, 1), blk, 0, stream>>>(cw2, cb2, t2b, sc2, sh2, wlb);

  // 8) softmax over k^2 + windowed aggregation with x3 (u3 never materialized)
  {
    long tot = (long)NN * GRP * HWSZ;
    softmax_agg_kernel<<<(unsigned)((tot + 255) / 256), 256, 0, stream>>>(wlb, x3b, out);
  }
}
